// RSAGE_Hetero_39633958208182
// MI455X (gfx1250) — matrix operand order repaired and checked by reading, not compiled, not run on hardware
//
#include <hip/hip_runtime.h>
#include <hip/hip_bf16.h>

typedef __attribute__((ext_vector_type(16))) _Float16 v16h;
typedef __attribute__((ext_vector_type(8)))  float    v8f;

#define NP_   50000
#define NA_   20000
#define D_IN  128
#define D_HID 128
#define D_OUTC 64

// ---------------------------------------------------------------------------
// Degree count: one thread per edge, atomic add 1.0 into cnt[dst].
// ---------------------------------------------------------------------------
__global__ void count_kernel(const int* __restrict__ dst, float* __restrict__ cnt, int E) {
    int e = blockIdx.x * blockDim.x + threadIdx.x;
    if (e < E) atomicAdd(cnt + dst[e], 1.0f);
}

// ---------------------------------------------------------------------------
// Scatter-add of 128-float rows: 32 threads per edge, float4 per thread.
// ---------------------------------------------------------------------------
__global__ void scatter_kernel(const float* __restrict__ x, const int* __restrict__ src,
                               const int* __restrict__ dst, float* __restrict__ agg, int E) {
    int t = blockIdx.x * blockDim.x + threadIdx.x;
    int e = t >> 5;
    if (e >= E) return;
    int f = (t & 31) << 2;
    const float4 v = *(const float4*)(x + (size_t)src[e] * 128 + f);
    float* o = agg + (size_t)dst[e] * 128 + f;
    atomicAdd(o + 0, v.x);
    atomicAdd(o + 1, v.y);
    atomicAdd(o + 2, v.z);
    atomicAdd(o + 3, v.w);
}

// ---------------------------------------------------------------------------
// agg[node] *= 1/max(cnt[node],1): 32 threads per node.
// ---------------------------------------------------------------------------
__global__ void normalize_kernel(float* __restrict__ agg, const float* __restrict__ cnt, int n) {
    int t = blockIdx.x * blockDim.x + threadIdx.x;
    int node = t >> 5;
    if (node >= n) return;
    int f = (t & 31) << 2;
    float s = 1.0f / fmaxf(cnt[node], 1.0f);
    float4* p = (float4*)(agg + (size_t)node * 128 + f);
    float4 v = *p;
    v.x *= s; v.y *= s; v.z *= s; v.w *= s;
    *p = v;
}

// ---------------------------------------------------------------------------
// Pack effective weight matrix into CDNA5 WMMA 16-bit B-fragment order.
// Logical B[k][n] = Wseg[n*128 + (k&127)] (Wl.T), k-segment 2 may be a sum.
// Fragment layout (ISA 7.12.2, 16-bit B 32x16): lane%16 = N, lane/16 = half,
// k_local = half*8 + e for e<8, 16 + half*8 + (e-8) for e>=8, packed 2 per dword.
// Linear index == (((c*NTILES + ntile)*32 + lane)*16 + e), matching the GEMM.
// ---------------------------------------------------------------------------
__global__ void pack_b_kernel(const float* __restrict__ W0, const float* __restrict__ W1,
                              const float* __restrict__ W2a, const float* __restrict__ W2b,
                              int nsegs, int N, float scale, _Float16* __restrict__ Bpack) {
    int idx = blockIdx.x * blockDim.x + threadIdx.x;
    int total = nsegs * 128 * N;
    if (idx >= total) return;
    int e      = idx & 15;
    int lane   = (idx >> 4) & 31;
    int rest   = idx >> 9;
    int ntiles = N >> 4;
    int ntile  = rest % ntiles;
    int c      = rest / ntiles;                 // K-chunk of 32
    int n      = ntile * 16 + (lane & 15);
    int k      = c * 32 + (lane >> 4) * 8 + (e & 7) + (e >> 3) * 16; // global K
    int seg    = k >> 7;
    int kk     = k & 127;
    float w;
    if (seg == 0)      w = W0[n * 128 + kk];
    else if (seg == 1) w = W1[n * 128 + kk];
    else {             w = W2a[n * 128 + kk];
                       if (W2b) w += W2b[n * 128 + kk]; }
    Bpack[idx] = (_Float16)(w * scale);
}

__global__ void pack_bias_kernel(const float* __restrict__ ba, const float* __restrict__ bb,
                                 float scale, float* __restrict__ beff, int N) {
    int n = blockIdx.x * blockDim.x + threadIdx.x;
    if (n >= N) return;
    float v = ba[n];
    if (bb) v += bb[n];
    beff[n] = v * scale;
}

// ---------------------------------------------------------------------------
// WMMA GEMM: out[M x N] = act( [A0|A1|A2] @ Bpacked + beff ).
// M-tile of 16 per block, 4 waves, NT n-tiles of 16 per wave (N = NT*64).
// A fragment per ISA 16-bit A 16x32 layout: lane%16 = M row, lane/16 = half,
// elements 0..7 -> K = half*8+e, elements 8..15 -> K = 16+half*8+(e-8).
// Accumulate fp32 via v_wmma_f32_16x16x32_f16.
// ---------------------------------------------------------------------------
template <int NSEG, int NT>
__global__ __launch_bounds__(128) void wmma_gemm_kernel(
    const float* __restrict__ A0, const float* __restrict__ A1, const float* __restrict__ A2,
    const _Float16* __restrict__ Bpack, const float* __restrict__ beff,
    float* __restrict__ out, int relu) {
    constexpr int N      = NT * 64;
    constexpr int NTILES = N / 16;
    const int mtile = blockIdx.x;
    const int wave  = threadIdx.x >> 5;
    const int lane  = threadIdx.x & 31;
    const int m     = lane & 15;
    const int half  = lane >> 4;
    const size_t rowBase = ((size_t)(mtile * 16 + m)) * 128;

    const float* Asrc[3] = {A0, A1, A2};
    v8f acc[NT] = {};

#pragma unroll
    for (int c = 0; c < NSEG * 4; ++c) {
        const float* Ab = Asrc[c >> 2] + rowBase + (c & 3) * 32;
        float4 q0 = *(const float4*)(Ab + (half << 3));
        float4 q1 = *(const float4*)(Ab + (half << 3) + 4);
        float4 q2 = *(const float4*)(Ab + 16 + (half << 3));
        float4 q3 = *(const float4*)(Ab + 16 + (half << 3) + 4);
        v16h a;
        a[0]  = (_Float16)q0.x; a[1]  = (_Float16)q0.y; a[2]  = (_Float16)q0.z; a[3]  = (_Float16)q0.w;
        a[4]  = (_Float16)q1.x; a[5]  = (_Float16)q1.y; a[6]  = (_Float16)q1.z; a[7]  = (_Float16)q1.w;
        a[8]  = (_Float16)q2.x; a[9]  = (_Float16)q2.y; a[10] = (_Float16)q2.z; a[11] = (_Float16)q2.w;
        a[12] = (_Float16)q3.x; a[13] = (_Float16)q3.y; a[14] = (_Float16)q3.z; a[15] = (_Float16)q3.w;
#pragma unroll
        for (int t = 0; t < NT; ++t) {
            int ntile = wave * NT + t;
            const _Float16* bp = Bpack + (((size_t)c * NTILES + ntile) * 32 + lane) * 16;
            v16h b = *(const v16h*)bp;
            acc[t] = __builtin_amdgcn_wmma_f32_16x16x32_f16(
                false, a, false, b, (short)0, acc[t], false, false);
        }
    }

#pragma unroll
    for (int t = 0; t < NT; ++t) {
        int ntile = wave * NT + t;
        int n = ntile * 16 + m;
        float bias = beff[n];
#pragma unroll
        for (int r = 0; r < 8; ++r) {
            int mm = mtile * 16 + r + half * 8;   // C/D layout: lanes 16-31 hold M+8
            float v = acc[t][r] + bias;
            if (relu) v = fmaxf(v, 0.0f);
            out[(size_t)mm * N + n] = v;
        }
    }
}

// ---------------------------------------------------------------------------
extern "C" void kernel_launch(void* const* d_in, const int* in_sizes, int n_in,
                              void* d_out, int out_size, void* d_ws, size_t ws_size,
                              hipStream_t stream) {
    const float* x_paper  = (const float*)d_in[0];
    const float* x_author = (const float*)d_in[1];
    const float* Wl1 = (const float*)d_in[2];   // (3,128,128)
    const float* bl1 = (const float*)d_in[3];   // (3,128)
    const float* Wr1 = (const float*)d_in[4];   // (3,128,128)
    const float* Wl2 = (const float*)d_in[5];   // (3,64,128)
    const float* bl2 = (const float*)d_in[6];   // (3,64)
    const float* Wr2 = (const float*)d_in[7];   // (3,64,128)
    const int* writes_src = (const int*)d_in[8];
    const int* writes_dst = (const int*)d_in[9];
    const int* cites_src  = (const int*)d_in[10];
    const int* cites_dst  = (const int*)d_in[11];
    const int* rev_src    = (const int*)d_in[12];
    const int* rev_dst    = (const int*)d_in[13];
    const int E_W = in_sizes[8];
    const int E_C = in_sizes[10];
    const int E_R = in_sizes[12];

    // ---- workspace carve-up (256B aligned) ----
    char* ws = (char*)d_ws;
    size_t off = 0;
    auto carve = [&](size_t bytes) -> char* {
        char* p = ws + off;
        off = (off + bytes + 255) & ~(size_t)255;
        return p;
    };
    float* aggW = (float*)carve((size_t)NP_ * 128 * 4);   // writes agg (papers)
    float* aggC = (float*)carve((size_t)NP_ * 128 * 4);   // cites agg (papers)
    float* aggR = (float*)carve((size_t)NA_ * 128 * 4);   // rev agg (authors)
    float* hp1  = (float*)carve((size_t)NP_ * 128 * 4);
    float* ha1  = (float*)carve((size_t)NA_ * 128 * 4);
    float* cntW = (float*)carve((size_t)NP_ * 4);
    float* cntC = (float*)carve((size_t)NP_ * 4);
    float* cntR = (float*)carve((size_t)NA_ * 4);
    _Float16* Bp1p = (_Float16*)carve((size_t)3 * 128 * 128 * 2); // K=384,N=128
    _Float16* Bp1a = (_Float16*)carve((size_t)2 * 128 * 128 * 2); // K=256,N=128
    _Float16* Bp2p = (_Float16*)carve((size_t)3 * 128 * 64 * 2);  // K=384,N=64
    float* be1p = (float*)carve(128 * 4);
    float* be1a = (float*)carve(128 * 4);
    float* be2p = (float*)carve(64 * 4);
    (void)ws_size; (void)n_in; (void)out_size;

    const int T = 256;
    auto gse = [&](long long n) { return (unsigned)((n + T - 1) / T); };

    // ---- zero counters + layer-1 aggregation buffers ----
    hipMemsetAsync(cntW, 0, (size_t)NP_ * 4, stream);
    hipMemsetAsync(cntC, 0, (size_t)NP_ * 4, stream);
    hipMemsetAsync(cntR, 0, (size_t)NA_ * 4, stream);
    hipMemsetAsync(aggW, 0, (size_t)NP_ * 128 * 4, stream);
    hipMemsetAsync(aggC, 0, (size_t)NP_ * 128 * 4, stream);
    hipMemsetAsync(aggR, 0, (size_t)NA_ * 128 * 4, stream);

    // ---- degree counts (layer-invariant) ----
    count_kernel<<<gse(E_W), T, 0, stream>>>(writes_dst, cntW, E_W);
    count_kernel<<<gse(E_C), T, 0, stream>>>(cites_dst, cntC, E_C);
    count_kernel<<<gse(E_R), T, 0, stream>>>(rev_dst, cntR, E_R);

    // ---- weight/bias packing (layer-invariant) ----
    pack_b_kernel<<<gse(3LL * 128 * 128), T, 0, stream>>>(
        Wl1, Wl1 + 16384, Wr1, Wr1 + 16384, 3, 128, 0.5f, Bp1p);
    pack_b_kernel<<<gse(2LL * 128 * 128), T, 0, stream>>>(
        Wl1 + 2 * 16384, Wr1 + 2 * 16384, nullptr, nullptr, 2, 128, 1.0f, Bp1a);
    pack_b_kernel<<<gse(3LL * 128 * 64), T, 0, stream>>>(
        Wl2, Wl2 + 8192, Wr2, Wr2 + 8192, 3, 64, 0.5f, Bp2p);
    pack_bias_kernel<<<1, 128, 0, stream>>>(bl1, bl1 + 128, 0.5f, be1p, 128);
    pack_bias_kernel<<<1, 128, 0, stream>>>(bl1 + 256, nullptr, 1.0f, be1a, 128);
    pack_bias_kernel<<<1, 64, 0, stream>>>(bl2, bl2 + 64, 0.5f, be2p, 64);

    // ---- layer 1: scatter + normalize ----
    scatter_kernel<<<gse(32LL * E_W), T, 0, stream>>>(x_author, writes_src, writes_dst, aggW, E_W);
    scatter_kernel<<<gse(32LL * E_C), T, 0, stream>>>(x_paper, cites_src, cites_dst, aggC, E_C);
    scatter_kernel<<<gse(32LL * E_R), T, 0, stream>>>(x_paper, rev_src, rev_dst, aggR, E_R);
    normalize_kernel<<<gse(32LL * NP_), T, 0, stream>>>(aggW, cntW, NP_);
    normalize_kernel<<<gse(32LL * NP_), T, 0, stream>>>(aggC, cntC, NP_);
    normalize_kernel<<<gse(32LL * NA_), T, 0, stream>>>(aggR, cntR, NA_);

    // ---- layer 1 GEMMs (WMMA) ----
    // hp1 = relu(0.5*(aggW@Wl1_0.T + aggC@Wl1_1.T + xp@(Wr1_0+Wr1_1).T) + 0.5*(bl1_0+bl1_1))
    wmma_gemm_kernel<3, 2><<<NP_ / 16, 128, 0, stream>>>(
        aggW, aggC, x_paper, Bp1p, be1p, hp1, 1);
    // ha1 = relu(aggR@Wl1_2.T + xa@Wr1_2.T + bl1_2)
    wmma_gemm_kernel<2, 2><<<NA_ / 16, 128, 0, stream>>>(
        aggR, x_author, nullptr, Bp1a, be1a, ha1, 1);

    // ---- layer 2: re-aggregate hidden features ----
    hipMemsetAsync(aggW, 0, (size_t)NP_ * 128 * 4, stream);
    hipMemsetAsync(aggC, 0, (size_t)NP_ * 128 * 4, stream);
    scatter_kernel<<<gse(32LL * E_W), T, 0, stream>>>(ha1, writes_src, writes_dst, aggW, E_W);
    scatter_kernel<<<gse(32LL * E_C), T, 0, stream>>>(hp1, cites_src, cites_dst, aggC, E_C);
    normalize_kernel<<<gse(32LL * NP_), T, 0, stream>>>(aggW, cntW, NP_);
    normalize_kernel<<<gse(32LL * NP_), T, 0, stream>>>(aggC, cntC, NP_);

    // ---- layer 2 GEMM -> d_out (float, NP x 64, no activation) ----
    wmma_gemm_kernel<3, 1><<<NP_ / 16, 128, 0, stream>>>(
        aggW, aggC, hp1, Bp2p, be2p, (float*)d_out, 0);
}
